// LeafOnlyNet_63376537420417
// MI455X (gfx1250) — matrix operand order repaired and checked by reading, not compiled, not run on hardware
//
#include <hip/hip_runtime.h>

typedef __attribute__((ext_vector_type(16))) __bf16 bf16x16;
typedef __attribute__((ext_vector_type(8)))  __bf16 bf16x8;
typedef __attribute__((ext_vector_type(8)))  float  f32x8;

constexpr int cB  = 4;
constexpr int cNB = 128;
constexpr int cL  = 64;
constexpr int cLQ = 65;          // L + 1
constexpr int cC  = 512;
constexpr int cH  = 8;
constexpr int cHD = 64;
constexpr int cM  = cB * cNB * cLQ;   // 33280 token rows
constexpr float cSCALE = 0.125f;      // HD^-0.5

__device__ __forceinline__ f32x8 wmma_bf16(bf16x16 a, bf16x16 b, f32x8 c) {
  return __builtin_amdgcn_wmma_f32_16x16x32_bf16(false, a, false, b,
                                                 (short)0, c, false, false);
}

__device__ __forceinline__ bf16x16 cat16(bf16x8 lo, bf16x8 hi) {
  return __builtin_shufflevector(lo, hi, 0, 1, 2, 3, 4, 5, 6, 7,
                                          8, 9, 10, 11, 12, 13, 14, 15);
}

// ---------------------------------------------------------------- converts
__global__ void cvt_bf16(const float* __restrict__ in, __bf16* __restrict__ out,
                         int n) {
  for (int i = blockIdx.x * blockDim.x + threadIdx.x; i < n;
       i += gridDim.x * blockDim.x)
    out[i] = (__bf16)in[i];
}

// ------------------------------------------------- 32x32-per-wave GEMM core
// Y[m][n] = sum_k A[m][k] * W[n][k]   (A row-major lda, W row-major ldb)
__device__ __forceinline__ void gemm_tile_32x32(
    const __bf16* __restrict__ A, const __bf16* __restrict__ W,
    int m0, int n0, int K, int lda, int ldb, int lane, f32x8 (&acc)[2][2]) {
  const int hi = lane >> 4, ln = lane & 15;
  for (int k0 = 0; k0 < K; k0 += 32) {
    bf16x16 av[2], bv[2];
#pragma unroll
    for (int t = 0; t < 2; ++t) {
      const __bf16* p = A + (size_t)(m0 + t * 16 + ln) * lda + k0 + hi * 8;
      bf16x8 lo = *(const bf16x8*)p;
      bf16x8 h8 = *(const bf16x8*)(p + 16);
      av[t] = cat16(lo, h8);
    }
#pragma unroll
    for (int u = 0; u < 2; ++u) {
      const __bf16* p = W + (size_t)(n0 + u * 16 + ln) * ldb + k0 + hi * 8;
      bf16x8 lo = *(const bf16x8*)p;
      bf16x8 h8 = *(const bf16x8*)(p + 16);
      bv[u] = cat16(lo, h8);
    }
#pragma unroll
    for (int t = 0; t < 2; ++t)
#pragma unroll
      for (int u = 0; u < 2; ++u)
        acc[t][u] = wmma_bf16(av[t], bv[u], acc[t][u]);
  }
}

// ------------------------------------------------------------- QKV GEMM
// out (cM x 1536) bf16 = xb (cM x 512) * qkv_w^T + qkv_b
__global__ __launch_bounds__(256) void qkv_gemm(
    const __bf16* __restrict__ xb, const __bf16* __restrict__ wb,
    const float* __restrict__ bias, __bf16* __restrict__ out) {
  const int wave = threadIdx.x >> 5, lane = threadIdx.x & 31;
  const int m0 = blockIdx.x * 64 + (wave >> 2) * 32;
  const int n0 = blockIdx.y * 128 + (wave & 3) * 32;
  f32x8 acc[2][2] = {};
  gemm_tile_32x32(xb, wb, m0, n0, cC, cC, cC, lane, acc);
  const int hi = lane >> 4, ln = lane & 15;
#pragma unroll
  for (int t = 0; t < 2; ++t)
#pragma unroll
    for (int u = 0; u < 2; ++u) {
      const int col = n0 + u * 16 + ln;
      const float bv = bias[col];
#pragma unroll
      for (int r = 0; r < 8; ++r) {
        const int row = m0 + t * 16 + 8 * hi + r;
        out[(size_t)row * (3 * cC) + col] = (__bf16)(acc[t][u][r] + bv);
      }
    }
}

// ------------------------------------------------------------- Out proj
// d_out (cM x 512) f32 = ao (cM x 512) * proj_w^T + proj_b
__global__ __launch_bounds__(256) void proj_gemm(
    const __bf16* __restrict__ ao, const __bf16* __restrict__ wb,
    const float* __restrict__ bias, float* __restrict__ out) {
  const int wave = threadIdx.x >> 5, lane = threadIdx.x & 31;
  const int m0 = blockIdx.x * 64 + (wave >> 2) * 32;
  const int n0 = blockIdx.y * 128 + (wave & 3) * 32;
  f32x8 acc[2][2] = {};
  gemm_tile_32x32(ao, wb, m0, n0, cC, cC, cC, lane, acc);
  const int hi = lane >> 4, ln = lane & 15;
#pragma unroll
  for (int t = 0; t < 2; ++t)
#pragma unroll
    for (int u = 0; u < 2; ++u) {
      const int col = n0 + u * 16 + ln;
      const float bv = bias[col];
#pragma unroll
      for (int r = 0; r < 8; ++r) {
        const int row = m0 + t * 16 + 8 * hi + r;
        out[(size_t)row * cC + col] = acc[t][u][r] + bv;
      }
    }
}

// -------------------------------------------------------- Attention kernel
// One block per (b, leaf, head). 5 waves, wave w handles q-rows [16w,16w+16).
__global__ __launch_bounds__(160) void attn_kernel(
    const __bf16* __restrict__ qkv,     // (cM, 1536) : [Q | K | V]
    const float*  __restrict__ edge,    // (B, NB, LQ, LQ, 4)
    const int*    __restrict__ mask,    // (B, NB, LQ, LQ)
    const float*  __restrict__ gate_w,  // (H, 4)
    const float*  __restrict__ gate_b,  // (H)
    __bf16*       __restrict__ ao) {    // (cM, 512)
  const int bid = blockIdx.x;                 // b*NB*H + nb*H + h
  const int h   = bid % cH;
  const int blk = bid / cH;                   // b*NB + nb  (0..511)
  const int rowbase = blk * cLQ;
  const int ldq = 3 * cC;

  const __bf16* Qp = qkv + (size_t)rowbase * ldq + h * cHD;
  const __bf16* Kp = Qp + cC;
  const __bf16* Vp = Qp + 2 * cC;

  __shared__ __attribute__((aligned(16))) __bf16 vt[cHD * 96];     // V^T padded
  __shared__ __attribute__((aligned(16))) __bf16 comb[5][16 * 96]; // per wave

  const int tid = threadIdx.x;
  const int wave = tid >> 5, lane = tid & 31;
  const int hi = lane >> 4, ln = lane & 15;
  const float NEG_INF = -__builtin_inff();

  // ---- stage V transposed into LDS (vt[hd][key]), zero-padded keys 65..95
  for (int i = tid; i < cHD * 96; i += 160) vt[i] = (__bf16)0.0f;
  __syncthreads();
  for (int i = tid; i < cLQ * cHD; i += 160) {
    const int r = i / cHD, c = i % cHD;   // key r, hd c
    vt[c * 96 + r] = Vp[(size_t)r * ldq + c];
  }
  __syncthreads();

  // ---- Q strip as A operand (16 x 64), rows >= 65 zeroed
  const int lq_a = wave * 16 + ln;
  const bool qa_ok = lq_a < cLQ;
  bf16x16 aq[2];
#pragma unroll
  for (int ks = 0; ks < 2; ++ks) {
    bf16x8 lo = {}, h8 = {};
    if (qa_ok) {
      const __bf16* p = Qp + (size_t)lq_a * ldq + ks * 32 + hi * 8;
      lo = *(const bf16x8*)p;
      h8 = *(const bf16x8*)(p + 16);
    }
    aq[ks] = cat16(lo, h8);
  }

  // ---- scores strip 16 x 80 : S = Q K^T
  f32x8 sc[5];
#pragma unroll
  for (int t = 0; t < 5; ++t) sc[t] = {};
#pragma unroll
  for (int nt = 0; nt < 5; ++nt) {
    const int key = nt * 16 + ln;
    const bool kok = key < cLQ;
#pragma unroll
    for (int ks = 0; ks < 2; ++ks) {
      bf16x8 lo = {}, h8 = {};
      if (kok) {
        const __bf16* p = Kp + (size_t)key * ldq + ks * 32 + hi * 8;
        lo = *(const bf16x8*)p;
        h8 = *(const bf16x8*)(p + 16);
      }
      sc[nt] = wmma_bf16(aq[ks], cat16(lo, h8), sc[nt]);
    }
  }

  // ---- scale + physics bias + mask; compute gate
  const float gw0 = gate_w[h * 4 + 0], gw1 = gate_w[h * 4 + 1];
  const float gw2 = gate_w[h * 4 + 2], gw3 = gate_w[h * 4 + 3];
  const float gbv = gate_b[h];
  const size_t mb = (size_t)blk * cLQ * cLQ;
  float gatev[5][8];
#pragma unroll
  for (int nt = 0; nt < 5; ++nt) {
    const int lk = nt * 16 + ln;
#pragma unroll
    for (int r = 0; r < 8; ++r) {
      const int lq = wave * 16 + 8 * hi + r;
      float s = sc[nt][r] * cSCALE;
      float g = 0.0f;
      if (lq < cLQ && lk < cLQ) {
        const int mk = mask[mb + (size_t)lq * cLQ + lk];
        float4 e = *(const float4*)(edge + (mb + (size_t)lq * cLQ + lk) * 4);
        if (lq == lk && lq < cL) { e.x = 0.f; e.y = 0.f; e.z = 0.f; e.w = 1.f; }
        s += e.w;
        if (mk == 0) {
          s = NEG_INF;
        } else {
          g = e.x * gw0 + e.y * gw1 + e.z * gw2 + e.w * gw3 + gbv;
        }
      } else {
        s = NEG_INF;
      }
      sc[nt][r] = s;
      gatev[nt][r] = g;
    }
  }

  // ---- softmax over keys (row lives in one 16-lane half), + gate
#pragma unroll
  for (int r = 0; r < 8; ++r) {
    float m = NEG_INF;
#pragma unroll
    for (int nt = 0; nt < 5; ++nt) m = fmaxf(m, sc[nt][r]);
    m = fmaxf(m, __shfl_xor(m, 1, 32));
    m = fmaxf(m, __shfl_xor(m, 2, 32));
    m = fmaxf(m, __shfl_xor(m, 4, 32));
    m = fmaxf(m, __shfl_xor(m, 8, 32));
    float sum = 0.0f;
#pragma unroll
    for (int nt = 0; nt < 5; ++nt) {
      const float e = __expf(sc[nt][r] - m);
      sc[nt][r] = e;
      sum += e;
    }
    sum += __shfl_xor(sum, 1, 32);
    sum += __shfl_xor(sum, 2, 32);
    sum += __shfl_xor(sum, 4, 32);
    sum += __shfl_xor(sum, 8, 32);
    const float inv = 1.0f / sum;
#pragma unroll
    for (int nt = 0; nt < 5; ++nt)
      sc[nt][r] = sc[nt][r] * inv + gatev[nt][r];
  }

  // ---- combined strip to LDS (bf16, A-operand-friendly row-major 16x96)
  __bf16* cs = &comb[wave][0];
#pragma unroll
  for (int nt = 0; nt < 5; ++nt)
#pragma unroll
    for (int r = 0; r < 8; ++r)
      cs[(8 * hi + r) * 96 + nt * 16 + ln] = (__bf16)sc[nt][r];
#pragma unroll
  for (int r = 0; r < 8; ++r)                 // zero pad cols 80..95
    cs[(8 * hi + r) * 96 + 80 + ln] = (__bf16)0.0f;
  __syncthreads();

  // ---- O = combined (16x96) @ V (96x64)
  f32x8 o[4];
#pragma unroll
  for (int nt = 0; nt < 4; ++nt) o[nt] = {};
#pragma unroll
  for (int kt = 0; kt < 3; ++kt) {
    const __bf16* pa = cs + ln * 96 + kt * 32 + hi * 8;
    bf16x16 a = cat16(*(const bf16x8*)pa, *(const bf16x8*)(pa + 16));
#pragma unroll
    for (int nt = 0; nt < 4; ++nt) {
      const __bf16* pv = &vt[(nt * 16 + ln) * 96 + kt * 32 + hi * 8];
      bf16x16 bvv = cat16(*(const bf16x8*)pv, *(const bf16x8*)(pv + 16));
      o[nt] = wmma_bf16(a, bvv, o[nt]);
    }
  }

  // ---- store (drop padded q rows)
#pragma unroll
  for (int nt = 0; nt < 4; ++nt) {
    const int hd = nt * 16 + ln;
#pragma unroll
    for (int r = 0; r < 8; ++r) {
      const int lq = wave * 16 + 8 * hi + r;
      if (lq < cLQ)
        ao[(size_t)(rowbase + lq) * cC + h * cHD + hd] = (__bf16)o[nt][r];
    }
  }
}

// ------------------------------------------------------------------ launch
extern "C" void kernel_launch(void* const* d_in, const int* in_sizes, int n_in,
                              void* d_out, int out_size, void* d_ws,
                              size_t ws_size, hipStream_t stream) {
  const float* x       = (const float*)d_in[0];
  const int*   amask   = (const int*)  d_in[1];
  const float* edge    = (const float*)d_in[2];
  const float* qkv_w   = (const float*)d_in[3];
  const float* qkv_b   = (const float*)d_in[4];
  const float* proj_w  = (const float*)d_in[5];
  const float* proj_b  = (const float*)d_in[6];
  const float* gate_w  = (const float*)d_in[7];
  const float* gate_b  = (const float*)d_in[8];
  float* out = (float*)d_out;

  char* ws = (char*)d_ws;
  size_t off = 0;
  auto wsalloc = [&](size_t bytes) -> void* {
    void* p = ws + off;
    off = (off + bytes + 255) & ~(size_t)255;
    return p;
  };
  __bf16* xb  = (__bf16*)wsalloc((size_t)cM * cC * 2);            // x bf16
  __bf16* wqk = (__bf16*)wsalloc((size_t)3 * cC * cC * 2);        // qkv_w bf16
  __bf16* wpj = (__bf16*)wsalloc((size_t)cC * cC * 2);            // proj_w bf16
  __bf16* qkv = (__bf16*)wsalloc((size_t)cM * 3 * cC * 2);        // qkv out
  __bf16* ao  = (__bf16*)wsalloc((size_t)cM * cC * 2);            // attn out

  // converts
  {
    int n = cM * cC;                       // 17,039,360
    cvt_bf16<<<(n + 1023) / 1024, 256, 0, stream>>>(x, xb, n);
    n = 3 * cC * cC;
    cvt_bf16<<<(n + 1023) / 1024, 256, 0, stream>>>(qkv_w, wqk, n);
    n = cC * cC;
    cvt_bf16<<<(n + 1023) / 1024, 256, 0, stream>>>(proj_w, wpj, n);
  }

  // QKV projection: (cM x 512) x (512 x 1536)
  qkv_gemm<<<dim3(cM / 64, (3 * cC) / 128), 256, 0, stream>>>(xb, wqk, qkv_b,
                                                              qkv);

  // attention: one block per (b, leaf, head)
  attn_kernel<<<cB * cNB * cH, 160, 0, stream>>>(qkv, edge, amask, gate_w,
                                                 gate_b, ao);

  // output projection: (cM x 512) x (512 x 512) -> f32 d_out
  proj_gemm<<<dim3(cM / 64, cC / 128), 256, 0, stream>>>(ao, wpj, proj_b, out);
}
